// TimeSeriesModel_50620484550813
// MI455X (gfx1250) — matrix operand order repaired and checked by reading, not compiled, not run on hardware
//
#include <hip/hip_runtime.h>

// ---------------------------------------------------------------------------
// Fused 2-layer per-channel LSTM for MI455X (gfx1250), wave32 + WMMA bf16.
// 128 persistent workgroups = 8 channels x 16 hidden-slices of 16.
// Weights live in LDS (bf16); h vectors are exchanged through L2 with an
// atomic-counter barrier per (channel, timestep, layer).
// ---------------------------------------------------------------------------

typedef __attribute__((ext_vector_type(16))) __bf16 v16bf;
typedef __attribute__((ext_vector_type(8)))  __bf16 v8bf;
typedef __attribute__((ext_vector_type(8)))  float  v8f;

constexpr int B_  = 32;    // batch
constexpr int T_  = 512;   // seq len
constexpr int C_  = 8;     // channels
constexpr int H_  = 256;   // hidden
constexpr int G_  = 1024;  // 4*H
constexpr int S_  = 16;    // slices per channel
constexpr int HS_ = 16;    // hidden indices per slice
constexpr int N_  = 64;    // gate columns per slice (4 gates x 16)
constexpr int HP_ = 264;   // padded bf16 row stride for h buffers and W1 (K=256)
constexpr int K2P_= 520;   // padded bf16 row stride for W2 (K=512)
constexpr int GSTR_=65;    // float row stride for gate staging

// LDS carve offsets (bytes), all multiples of 16
constexpr size_t OFF_W1 = 0;
constexpr size_t OFF_W2 = OFF_W1 + (size_t)N_ * HP_  * 2;  //  33792
constexpr size_t OFF_H1 = OFF_W2 + (size_t)N_ * K2P_ * 2;  // +66560
constexpr size_t OFF_H2 = OFF_H1 + (size_t)B_ * HP_  * 2;  // +16896
constexpr size_t OFF_G  = OFF_H2 + (size_t)B_ * HP_  * 2;  // +16896
constexpr size_t OFF_C1 = OFF_G  + (size_t)B_ * GSTR_* 4;  // + 8320
constexpr size_t OFF_C2 = OFF_C1 + (size_t)B_ * HS_  * 4;  // + 2048
constexpr size_t OFF_WI = OFF_C2 + (size_t)B_ * HS_  * 4;  // + 2048
constexpr size_t OFF_B1 = OFF_WI + (size_t)N_ * 4;         // +  256
constexpr size_t OFF_B2 = OFF_B1 + (size_t)N_ * 4;         // +  256
constexpr size_t OFF_WL = OFF_B2 + (size_t)N_ * 4;         // +  256
constexpr size_t OFF_XS = OFF_WL + (size_t)H_ * 4;         // + 1024
constexpr size_t SMEM_TOTAL = OFF_XS + (size_t)B_ * 4;     // = 148480 bytes
static_assert(SMEM_TOTAL <= 320u * 1024u, "fits in 320KB WGP LDS");

static __device__ __forceinline__ float bf2f(__bf16 h) {
  unsigned short s = __builtin_bit_cast(unsigned short, h);
  unsigned u = ((unsigned)s) << 16;
  return __builtin_bit_cast(float, u);
}
static __device__ __forceinline__ __bf16 f2bf(float f) {
  unsigned u = __builtin_bit_cast(unsigned, f);
  unsigned r = (u + 0x7FFFu + ((u >> 16) & 1u)) >> 16;   // RNE
  return __builtin_bit_cast(__bf16, (unsigned short)r);
}
static __device__ __forceinline__ float sigm(float x) { return 1.f / (1.f + __expf(-x)); }
static __device__ __forceinline__ float tanh_f(float x) {
  float e = __expf(-2.f * x);
  return (1.f - e) / (1.f + e);
}

// A operand (16x32 bf16, M x K): lane L holds row M=L%16; K chunks
// [kb + (L<16?0:8), +8) in elems 0..7 and [kb+16+(L<16?0:8), +8) in elems 8..15.
static __device__ __forceinline__ v16bf ldA(const __bf16* p) {
  v8bf lo = *(const v8bf*)p;
  v8bf hi = *(const v8bf*)(p + 16);
  v16bf r;
#pragma unroll
  for (int i = 0; i < 8; ++i) { r[i] = lo[i]; r[i + 8] = hi[i]; }
  return r;
}
// B operand (32x16 bf16, K x N): lane L holds col N=L%16; K chunks
// [kb + (L<16?0:8), +8) in elems 0..7 and [kb+16+(L<16?0:8), +8) in elems 8..15.  B[k][n] = W[row n][k]  (gates = h * W^T).
static __device__ __forceinline__ v16bf ldB(const __bf16* p) {
  v8bf lo = *(const v8bf*)p;
  v8bf hi = *(const v8bf*)(p + 16);
  v16bf r;
#pragma unroll
  for (int i = 0; i < 8; ++i) { r[i] = lo[i]; r[i + 8] = hi[i]; }
  return r;
}

// device-scope barrier across the 16 slice-workgroups of one channel
static __device__ __forceinline__ void xbar(unsigned* flag, int tid) {
  __threadfence();
  __syncthreads();
  if (tid == 0) {
    __hip_atomic_fetch_add(flag, 1u, __ATOMIC_ACQ_REL, __HIP_MEMORY_SCOPE_AGENT);
    while (__hip_atomic_load(flag, __ATOMIC_ACQUIRE, __HIP_MEMORY_SCOPE_AGENT) < (unsigned)S_)
      __builtin_amdgcn_s_sleep(8);
  }
  __syncthreads();
  __threadfence();   // invalidate stale lines on every wave before the gather
}

__global__ void __launch_bounds__(256)
lstm_fused_kernel(const float* __restrict__ x,
                  const float* __restrict__ Wih1, const float* __restrict__ Whh1,
                  const float* __restrict__ bih1, const float* __restrict__ bhh1,
                  const float* __restrict__ Wih2, const float* __restrict__ Whh2,
                  const float* __restrict__ bih2, const float* __restrict__ bhh2,
                  const float* __restrict__ Wlin, const float* __restrict__ blin,
                  float* __restrict__ out,
                  __bf16* __restrict__ Hsh1, __bf16* __restrict__ Hsh2,
                  unsigned* __restrict__ cnt) {
  const int tid = threadIdx.x;
  const int c   = blockIdx.x & (C_ - 1);
  const int s   = blockIdx.x >> 3;
  const int hs  = s * HS_;

  extern __shared__ __align__(16) char smem[];
  __bf16* W1    = (__bf16*)(smem + OFF_W1);   // [64][264]
  __bf16* W2    = (__bf16*)(smem + OFF_W2);   // [64][520]  (Wih2 | Whh2)
  __bf16* h1f   = (__bf16*)(smem + OFF_H1);   // [32][264]
  __bf16* h2f   = (__bf16*)(smem + OFF_H2);   // [32][264]
  float*  gates = (float*)(smem + OFF_G);     // [32][65]
  float*  c1s   = (float*)(smem + OFF_C1);    // [32][16]
  float*  c2s   = (float*)(smem + OFF_C2);    // [32][16]
  float*  wih1s = (float*)(smem + OFF_WI);    // [64]
  float*  b1s   = (float*)(smem + OFF_B1);    // [64]
  float*  b2s   = (float*)(smem + OFF_B2);    // [64]
  float*  wlins = (float*)(smem + OFF_WL);    // [256]
  float*  xs    = (float*)(smem + OFF_XS);    // [32]

  // ---------------- preload weights (f32 -> bf16 LDS) ----------------
  for (int idx = tid; idx < N_ * H_; idx += 256) {
    int n = idx >> 8, k = idx & 255;
    int gate = n >> 4, hh = n & 15;
    int grow = gate * H_ + hs + hh;
    W1[n * HP_ + k] = f2bf(Whh1[((size_t)c * G_ + grow) * H_ + k]);
  }
  for (int idx = tid; idx < N_ * 512; idx += 256) {
    int n = idx >> 9, k = idx & 511;
    int gate = n >> 4, hh = n & 15;
    int grow = gate * H_ + hs + hh;
    float w = (k < H_) ? Wih2[((size_t)c * G_ + grow) * H_ + k]
                       : Whh2[((size_t)c * G_ + grow) * H_ + (k - H_)];
    W2[n * K2P_ + k] = f2bf(w);
  }
  if (tid < N_) {
    int gate = tid >> 4, hh = tid & 15;
    int grow = gate * H_ + hs + hh;
    wih1s[tid] = Wih1[(size_t)c * G_ + grow];
    b1s[tid] = bih1[(size_t)c * G_ + grow] + bhh1[(size_t)c * G_ + grow];
    b2s[tid] = bih2[(size_t)c * G_ + grow] + bhh2[(size_t)c * G_ + grow];
  }
  wlins[tid] = Wlin[(size_t)c * H_ + tid];
  for (int idx = tid; idx < B_ * HP_; idx += 256) { h1f[idx] = f2bf(0.f); h2f[idx] = f2bf(0.f); }
  for (int idx = tid; idx < B_ * HS_; idx += 256) { c1s[idx] = 0.f; c2s[idx] = 0.f; }
  __syncthreads();

  // wave -> one 16x16 output tile: mt over B (2 tiles), nt over 64 gate cols (4 tiles)
  const int lane = tid & 31;
  const int wv   = tid >> 5;
  const int mt   = wv & 1;
  const int nt   = wv >> 1;
  const int arow = mt * 16 + (lane & 15);
  const int brow = nt * 16 + (lane & 15);
  const int klo  = (lane & 16) ? 8 : 0;
  const int kbo  = (lane & 16) ? 8 : 0;
  const int mbase = mt * 16 + ((lane & 16) ? 8 : 0);
  const int ncol  = nt * 16 + (lane & 15);

  const __bf16* a1base = h1f + arow * HP_ + klo;
  const __bf16* a2base = h2f + arow * HP_ + klo;
  const __bf16* b1base = W1 + brow * HP_ + kbo;
  const __bf16* b2base = W2 + brow * K2P_ + kbo;

  for (int t = 0; t < T_; ++t) {
    // ================= layer 1: gates = h1_{t-1} @ Whh1_slice^T =================
    v8f acc = {0.f, 0.f, 0.f, 0.f, 0.f, 0.f, 0.f, 0.f};
#pragma unroll
    for (int kk = 0; kk < 8; ++kk) {
      v16bf a = ldA(a1base + kk * 32);
      v16bf b = ldB(b1base + kk * 32);
      acc = __builtin_amdgcn_wmma_f32_16x16x32_bf16(false, a, false, b, (short)0, acc,
                                                    false, false);
    }
#pragma unroll
    for (int r = 0; r < 8; ++r) gates[(mbase + r) * GSTR_ + ncol] = acc[r];
    if (tid < B_) xs[tid] = x[((size_t)tid * T_ + t) * C_ + c];
    __syncthreads();

#pragma unroll
    for (int p = 0; p < 2; ++p) {
      int idx = p * 256 + tid;          // 512 = 32 b x 16 hh
      int b = idx >> 4, hh = idx & 15;
      float xi = xs[b];
      const float* g = gates + b * GSTR_;
      float gi = g[hh]      + xi * wih1s[hh]      + b1s[hh];
      float gf = g[16 + hh] + xi * wih1s[16 + hh] + b1s[16 + hh];
      float gg = g[32 + hh] + xi * wih1s[32 + hh] + b1s[32 + hh];
      float go = g[48 + hh] + xi * wih1s[48 + hh] + b1s[48 + hh];
      float iv = sigm(gi), fv = sigm(gf), gv = tanh_f(gg), ov = sigm(go);
      float cn = fv * c1s[b * HS_ + hh] + iv * gv;
      c1s[b * HS_ + hh] = cn;
      Hsh1[((size_t)c * B_ + b) * H_ + hs + hh] = f2bf(ov * tanh_f(cn));
    }
    xbar(&cnt[((size_t)c * T_ + t) * 2 + 0], tid);
    {  // gather full h1_t (16KB via uint4)
      const uint4* src = (const uint4*)(Hsh1 + (size_t)c * B_ * H_);
#pragma unroll
      for (int q = 0; q < 4; ++q) {
        int idx = q * 256 + tid;            // 1024 x 16B
        int b = idx >> 5, k8 = idx & 31;
        *(uint4*)(h1f + b * HP_ + k8 * 8) = src[idx];
      }
    }
    __syncthreads();

    // ====== layer 2: gates = [h1_t ; h2_{t-1}] @ [Wih2 ; Whh2]_slice^T ======
    v8f acc2 = {0.f, 0.f, 0.f, 0.f, 0.f, 0.f, 0.f, 0.f};
#pragma unroll
    for (int kk = 0; kk < 16; ++kk) {
      const __bf16* ap = (kk < 8) ? (a1base + kk * 32) : (a2base + (kk - 8) * 32);
      v16bf a = ldA(ap);
      v16bf b = ldB(b2base + kk * 32);
      acc2 = __builtin_amdgcn_wmma_f32_16x16x32_bf16(false, a, false, b, (short)0, acc2,
                                                     false, false);
    }
#pragma unroll
    for (int r = 0; r < 8; ++r) gates[(mbase + r) * GSTR_ + ncol] = acc2[r];
    __syncthreads();

#pragma unroll
    for (int p = 0; p < 2; ++p) {
      int idx = p * 256 + tid;
      int b = idx >> 4, hh = idx & 15;
      const float* g = gates + b * GSTR_;
      float gi = g[hh] + b2s[hh];
      float gf = g[16 + hh] + b2s[16 + hh];
      float gg = g[32 + hh] + b2s[32 + hh];
      float go = g[48 + hh] + b2s[48 + hh];
      float iv = sigm(gi), fv = sigm(gf), gv = tanh_f(gg), ov = sigm(go);
      float cn = fv * c2s[b * HS_ + hh] + iv * gv;
      c2s[b * HS_ + hh] = cn;
      Hsh2[((size_t)c * B_ + b) * H_ + hs + hh] = f2bf(ov * tanh_f(cn));
    }
    xbar(&cnt[((size_t)c * T_ + t) * 2 + 1], tid);
    {  // gather full h2_t
      const uint4* src = (const uint4*)(Hsh2 + (size_t)c * B_ * H_);
#pragma unroll
      for (int q = 0; q < 4; ++q) {
        int idx = q * 256 + tid;
        int b = idx >> 5, k8 = idx & 31;
        *(uint4*)(h2f + b * HP_ + k8 * 8) = src[idx];
      }
    }
    __syncthreads();

    // ============ output: out[b,t,c] = h2_t . Wlin[c] + blin[c] ============
    if (s == 0) {
      int b = tid >> 3, part = tid & 7;
      float ps = 0.f;
      const __bf16* hr = h2f + b * HP_ + part * 32;
      const float* wr = wlins + part * 32;
#pragma unroll
      for (int k = 0; k < 32; ++k) ps += bf2f(hr[k]) * wr[k];
      gates[b * GSTR_ + part] = ps;   // reuse gate staging as reduce scratch
    }
    __syncthreads();
    if (s == 0 && tid < B_) {
      float r = blin[c];
#pragma unroll
      for (int p = 0; p < 8; ++p) r += gates[tid * GSTR_ + p];
      out[((size_t)tid * T_ + t) * C_ + c] = r;
    }
    __syncthreads();   // protect gates before next step's GEMM store
  }
}

extern "C" void kernel_launch(void* const* d_in, const int* in_sizes, int n_in,
                              void* d_out, int out_size, void* d_ws, size_t ws_size,
                              hipStream_t stream) {
  const float* x    = (const float*)d_in[0];
  const float* Wih1 = (const float*)d_in[1];
  const float* Whh1 = (const float*)d_in[2];
  const float* bih1 = (const float*)d_in[3];
  const float* bhh1 = (const float*)d_in[4];
  const float* Wih2 = (const float*)d_in[5];
  const float* Whh2 = (const float*)d_in[6];
  const float* bih2 = (const float*)d_in[7];
  const float* bhh2 = (const float*)d_in[8];
  const float* Wlin = (const float*)d_in[9];
  const float* blin = (const float*)d_in[10];
  float* out = (float*)d_out;

  // workspace: [counters | Hsh1 bf16 | Hsh2 bf16]  (~290 KB)
  unsigned* cnt = (unsigned*)d_ws;
  const size_t cntBytes = (size_t)C_ * T_ * 2 * sizeof(unsigned);
  __bf16* Hsh1 = (__bf16*)((char*)d_ws + cntBytes);
  __bf16* Hsh2 = Hsh1 + (size_t)C_ * B_ * H_;

  hipMemsetAsync(d_ws, 0, cntBytes, stream);  // reset per-step barriers each call

  hipFuncSetAttribute(reinterpret_cast<const void*>(lstm_fused_kernel),
                      hipFuncAttributeMaxDynamicSharedMemorySize, (int)SMEM_TOTAL);

  lstm_fused_kernel<<<dim3(C_ * S_), dim3(256), SMEM_TOTAL, stream>>>(
      x, Wih1, Whh1, bih1, bhh1, Wih2, Whh2, bih2, bhh2, Wlin, blin,
      out, Hsh1, Hsh2, cnt);
}
